// RegionProposalNetwork_28192165331273
// MI455X (gfx1250) — matrix operand order repaired and checked by reading, not compiled, not run on hardware
//
#include <hip/hip_runtime.h>
#include <stdint.h>

// ---------------- problem constants ----------------
#define B_         8
#define A_TOTAL_   196416
#define K_TOT_     4576
#define POST_      1000
#define SORT_N_    8192
#define NSEG_      (K_TOT_ / 16)            // 286 16-element segments
#define NTILE_     ((K_TOT_ + 255) / 256)   // 18 WMMA scan tiles (16 segments each)
#define NMS_T_     0.7f
#define MIN_SIZE_  1.0f
#define IMG_W_     1024.0f
#define IMG_H_     768.0f
#define BBOX_CLIP_ 4.135166556742356f   // log(1000/16)

__constant__ int c_LN[5] = {147456, 36864, 9216, 2304, 576};     // per-level counts
__constant__ int c_LO[5] = {0, 147456, 184320, 193536, 195840};  // per-level offsets
__constant__ int c_LK[5] = {1000, 1000, 1000, 1000, 576};        // per-level k
__constant__ int c_KB[5] = {0, 1000, 2000, 3000, 4000};          // slot bases

// order-preserving float32 -> uint32 key (larger key == larger float)
__device__ __forceinline__ uint32_t fkey(float x) {
  uint32_t u = __float_as_uint(x);
  return (u & 0x80000000u) ? ~u : (u | 0x80000000u);
}

// ---------------- CDNA5 WMMA types ----------------
typedef _Float16 v16h __attribute__((ext_vector_type(16)));
typedef float    v8f  __attribute__((ext_vector_type(8)));

// ---------------- CDNA5 Tensor Data Mover: 1D contiguous global -> LDS ----------------
typedef unsigned int uv4 __attribute__((ext_vector_type(4)));
typedef int          iv4 __attribute__((ext_vector_type(4)));
typedef int          iv8 __attribute__((ext_vector_type(8)));

// Copy nbytes (multiple of 4, nbytes/4 <= 65535 elements) from gptr to LDS offset.
// Issued by one wave; caller must barrier before the WG reads the LDS tile.
__device__ __forceinline__ void tdm_load_1d(uint32_t lds_off, const void* gptr, uint32_t nbytes) {
#if defined(__HIP_DEVICE_COMPILE__)
  uint64_t ga    = (uint64_t)(uintptr_t)gptr;
  uint32_t elems = nbytes >> 2;            // 4-byte elements (data_size=2)
  uv4 g0;
  g0[0] = 1u;                                                   // count=1, is_restore=0, no gather
  g0[1] = lds_off;                                              // lds_addr [63:32]
  g0[2] = (uint32_t)(ga & 0xFFFFFFFFu);                         // global_addr[31:0]
  g0[3] = (uint32_t)((ga >> 32) & 0x01FFFFFFu) | (2u << 30);    // global_addr[56:32] | type=2
  iv8 g1;
  g1[0] = (int)(2u << 16);                                      // wg_mask=0, data_size=2 (4B)
  g1[1] = (int)((elems & 0xFFFFu) << 16);                       // tensor_dim0[15:0] @ [31:16]
  g1[2] = (int)(((elems >> 16) & 0xFFFFu) | (1u << 16));        // tensor_dim0[31:16], tensor_dim1=1
  g1[3] = (int)((elems & 0xFFFFu) << 16);                       // tensor_dim1 hi=0, tile_dim0=elems
  g1[4] = 1;                                                    // tile_dim1=1, tile_dim2=0
  g1[5] = (int)elems;                                           // tensor_dim0_stride[31:0]
  g1[6] = (int)(((elems >> 16) & 0xFFFFu) | ((elems & 0xFFFFu) << 16)); // d0_stride hi, d1_stride lo
  g1[7] = 0;
  iv4 z4 = {0, 0, 0, 0};
#if defined(__clang_major__) && (__clang_major__ >= 23)
  iv8 z8 = {0, 0, 0, 0, 0, 0, 0, 0};
  __builtin_amdgcn_tensor_load_to_lds(g0, g1, z4, z4, z8, 0);
#else
  __builtin_amdgcn_tensor_load_to_lds(g0, g1, z4, z4, 0);
#endif
  __builtin_amdgcn_s_wait_tensorcnt(0);
#else
  (void)lds_off; (void)gptr; (void)nbytes;
#endif
}

// ---------------- K0: init per-image max-coordinate accumulators ----------------
__global__ void k0_init(uint32_t* maxBits) {
  int i = threadIdx.x;
  if (i < B_) maxBits[i] = 0u;  // coords >= 0, so uint bits of 0.0f
}

// ---------------- K1: per (image,level) radix top-k select + decode ----------------
// 40 blocks (b = blk/5, lvl = blk%5), 256 threads.
__global__ void k1_select_decode(const float* __restrict__ objectness,
                                 const float* __restrict__ deltas,
                                 const float* __restrict__ anchors,
                                 float* __restrict__ candScore,
                                 uint32_t* __restrict__ candIdx,
                                 float4* __restrict__ candBox,
                                 uint32_t* __restrict__ maxBits) {
  __shared__ uint32_t hist[256];
  __shared__ uint32_t sh_prefix, sh_krem, sh_sel, sh_eq;

  const int b   = blockIdx.x / 5;
  const int lvl = blockIdx.x % 5;
  const int tid = threadIdx.x;
  const int nt  = blockDim.x;
  const int n   = c_LN[lvl];
  const int off = c_LO[lvl];
  const int k   = c_LK[lvl];
  const float* obj = objectness + (size_t)b * A_TOTAL_ + off;

  // ---- exact k-th largest via 4x8-bit MSB radix select over ordered keys ----
  uint32_t prefix = 0, prefmask = 0, krem = (uint32_t)k;
  for (int pass = 0; pass < 4; ++pass) {
    const int shift = 24 - 8 * pass;
    for (int h = tid; h < 256; h += nt) hist[h] = 0;
    __syncthreads();
    for (int i = tid; i < n; i += nt) {
      __builtin_prefetch(&obj[i + 8 * nt], 0, 0);  // global_prefetch_b8
      uint32_t key = fkey(obj[i]);
      if ((key & prefmask) == prefix) atomicAdd(&hist[(key >> shift) & 255u], 1u);
    }
    __syncthreads();
    if (tid == 0) {
      uint32_t kr = krem, d = 0;
      for (int dd = 255; dd >= 0; --dd) {
        uint32_t c = hist[dd];
        if (kr <= c) { d = (uint32_t)dd; break; }
        kr -= c;
      }
      sh_prefix = prefix | (d << shift);
      sh_krem   = kr;
    }
    __syncthreads();
    prefix   = sh_prefix;
    krem     = sh_krem;
    prefmask |= (0xFFu << shift);
    __syncthreads();
  }
  const uint32_t T = prefix;        // exact k-th largest key
  const uint32_t needEq = krem;     // #elements == T to take

  if (tid == 0) { sh_sel = (uint32_t)c_KB[lvl]; sh_eq = 0; }
  __syncthreads();

  // ---- select (key > T, plus needEq ties) and decode only the winners ----
  for (int i = tid; i < n; i += nt) {
    const float o = obj[i];
    const uint32_t key = fkey(o);
    int slot = -1;
    if (key > T) {
      slot = (int)atomicAdd(&sh_sel, 1u);
    } else if (key == T) {
      uint32_t e = atomicAdd(&sh_eq, 1u);
      if (e < needEq) slot = (int)atomicAdd(&sh_sel, 1u);
    }
    if (slot >= 0) {
      const int a = off + i;  // global anchor index
      const float4 an = ((const float4*)anchors)[a];
      const float4 dl = ((const float4*)deltas)[(size_t)b * A_TOTAL_ + a];
      const float wa = an.z - an.x, ha = an.w - an.y;
      const float cxa = an.x + 0.5f * wa, cya = an.y + 0.5f * ha;
      const float dw = fminf(dl.z, BBOX_CLIP_), dh = fminf(dl.w, BBOX_CLIP_);
      const float pcx = dl.x * wa + cxa, pcy = dl.y * ha + cya;
      const float pw = expf(dw) * wa, ph = expf(dh) * ha;
      float x1 = fminf(fmaxf(pcx - 0.5f * pw, 0.0f), IMG_W_);
      float y1 = fminf(fmaxf(pcy - 0.5f * ph, 0.0f), IMG_H_);
      float x2 = fminf(fmaxf(pcx + 0.5f * pw, 0.0f), IMG_W_);
      float y2 = fminf(fmaxf(pcy + 0.5f * ph, 0.0f), IMG_H_);
      const float sig = 1.0f / (1.0f + expf(-o));
      const bool valid = ((x2 - x1) >= MIN_SIZE_) && ((y2 - y1) >= MIN_SIZE_) && (sig >= 0.0f);
      const size_t ci = (size_t)b * K_TOT_ + (size_t)slot;
      candScore[ci] = valid ? sig : -1.0f;
      candIdx[ci]   = (uint32_t)a;
      candBox[ci]   = make_float4(x1, y1, x2, y2);
      atomicMax(&maxBits[b], __float_as_uint(x2));  // coords >= 0: uint-bit max == float max
      atomicMax(&maxBits[b], __float_as_uint(y2));
    }
  }
}

// ---------------- K2: per-image bitonic sort (desc) on composite 64-bit keys ----------------
// 8 blocks, 1024 threads. Scores/indices staged into LDS by the Tensor Data Mover.
// Key = score desc, anchorIdx asc (reference tie order), slot payload in low 13 bits.
__global__ void k2_sort(const float* __restrict__ candScore,
                        const uint32_t* __restrict__ candIdx,
                        uint32_t* __restrict__ order) {
  extern __shared__ __align__(16) char smem[];
  unsigned long long* keys = (unsigned long long*)smem;            // [SORT_N_]  65536 B
  float*    scL = (float*)(smem + SORT_N_ * 8);                    // [K_TOT_]   18304 B
  uint32_t* aiL = (uint32_t*)(smem + SORT_N_ * 8 + K_TOT_ * 4);    // [K_TOT_]   18304 B

  const int b = blockIdx.x, tid = threadIdx.x, nt = blockDim.x;

  if (tid == 0) {  // TDM DMA of the contiguous per-image slices into LDS
    tdm_load_1d((uint32_t)(uintptr_t)scL, &candScore[(size_t)b * K_TOT_], K_TOT_ * 4);
    tdm_load_1d((uint32_t)(uintptr_t)aiL, &candIdx[(size_t)b * K_TOT_],  K_TOT_ * 4);
  }
  __syncthreads();

  for (int j = tid; j < SORT_N_; j += nt) {
    unsigned long long key;
    if (j < K_TOT_) {
      const uint32_t a = aiL[j];
      key = ((unsigned long long)fkey(scL[j]) << 32)
          | ((unsigned long long)((~a) & 0x3FFFFu) << 13)   // smaller anchorIdx -> larger key
          | (unsigned long long)(uint32_t)j;                // slot payload (13 bits)
    } else {
      key = (unsigned long long)(uint32_t)j;                // tiny: sinks to the end
    }
    keys[j] = key;
  }
  __syncthreads();

  for (int k = 2; k <= SORT_N_; k <<= 1) {
    for (int j = k >> 1; j > 0; j >>= 1) {
      for (int i = tid; i < SORT_N_; i += nt) {
        const int ixj = i ^ j;
        if (ixj > i) {
          const unsigned long long a = keys[i], c = keys[ixj];
          const bool descSeg = ((i & k) == 0);
          const bool doSwap = descSeg ? (a < c) : (a > c);  // overall descending
          if (doSwap) { keys[i] = c; keys[ixj] = a; }
        }
      }
      __syncthreads();
    }
  }
  for (int j = tid; j < K_TOT_; j += nt)
    order[(size_t)b * K_TOT_ + j] = (uint32_t)(keys[j] & 0x1FFFu);
}

// ---------------- K3: materialize sorted, level-offset boxes (contiguous for TDM) ----------------
__global__ void k3_materialize(const uint32_t* __restrict__ order,
                               const float4* __restrict__ candBox,
                               const uint32_t* __restrict__ maxBits,
                               float4* __restrict__ sortedOff) {
  const int b = blockIdx.y;
  const int j = blockIdx.x * blockDim.x + threadIdx.x;
  if (j >= K_TOT_) return;
  const uint32_t slot = order[(size_t)b * K_TOT_ + j];
  float4 bx = candBox[(size_t)b * K_TOT_ + slot];
  int lvl = (int)(slot / 1000u); if (lvl > 4) lvl = 4;
  const float M = __uint_as_float(maxBits[b]) + 1.0f;   // jnp.max(boxes) + 1
  const float o = (float)lvl * M;
  bx.x += o; bx.y += o; bx.z += o; bx.w += o;
  sortedOff[(size_t)b * K_TOT_ + j] = bx;
}

// ---------------- K4: greedy NMS in LDS (boxes via TDM) + WMMA rank scan + output ----------------
// 8 blocks, 1024 threads (32 waves).
// LDS: boxes 73216 | area 18304 | keep 18304 | rank 18432 | segOff 1152 | inv 4000 = 133,408 B.
__global__ void k4_nms_out(const uint32_t* __restrict__ order,
                           const float* __restrict__ candScore,
                           const float4* __restrict__ candBox,
                           const float4* __restrict__ sortedOff,
                           float* __restrict__ out) {
  extern __shared__ __align__(16) char smem[];
  float4* boxs   = (float4*)smem;                                   // [K_TOT_]
  float*  area   = (float*)(smem + (size_t)K_TOT_ * 16);            // [K_TOT_]
  int*    keep   = (int*)(smem + (size_t)K_TOT_ * 20);              // [K_TOT_]
  int*    rank   = (int*)(smem + (size_t)K_TOT_ * 24);              // [NTILE_*256]
  int*    segOff = (int*)(smem + (size_t)K_TOT_ * 24 + NTILE_ * 1024); // [288]
  int*    inv    = (int*)(smem + (size_t)K_TOT_ * 24 + NTILE_ * 1024 + 288 * 4); // [POST_]

  const int b = blockIdx.x, tid = threadIdx.x, nt = blockDim.x;
  const int wv = tid >> 5, ln = tid & 31;

  // DMA the 73KB sorted box tile into LDS in a single tensor op (reused K^2 times).
  if (tid == 0)
    tdm_load_1d((uint32_t)(uintptr_t)boxs, &sortedOff[(size_t)b * K_TOT_],
                (uint32_t)(K_TOT_ * 16));
  __syncthreads();

  for (int i = tid; i < K_TOT_; i += nt) {
    const float4 bx = boxs[i];
    area[i] = (bx.z - bx.x) * (bx.w - bx.y);
    const float s = candScore[(size_t)b * K_TOT_ + order[(size_t)b * K_TOT_ + i]];
    keep[i] = (s >= 0.0f) ? 1 : 0;   // keep init = valid (invalid have s == -1)
  }
  for (int r = tid; r < POST_; r += nt) inv[r] = -1;
  __syncthreads();

  // Greedy scan matching the reference's sequential suppression semantics.
  int i = 0;
  while (i < K_TOT_) {
    while (i < K_TOT_ && keep[i] == 0) ++i;   // uniform skip over suppressed pivots
    if (i >= K_TOT_) break;
    const float4 bi = boxs[i];
    const float ai = area[i];
    for (int j = i + 1 + tid; j < K_TOT_; j += nt) {
      if (keep[j]) {
        const float4 bj = boxs[j];
        const float lx = fmaxf(bi.x, bj.x), ly = fmaxf(bi.y, bj.y);
        const float rx = fminf(bi.z, bj.z), ry = fminf(bi.w, bj.w);
        const float w = fmaxf(rx - lx, 0.0f), h = fmaxf(ry - ly, 0.0f);
        const float inter = w * h;
        const float iou = inter / fmaxf(ai + area[j] - inter, 1e-9f);
        if (iou > NMS_T_) keep[j] = 0;
      }
    }
    ++i;
    __syncthreads();
  }

  // ---- WMMA scan: D = L * X (f16 in, f32 acc; exact for 0/1 flags, sums <= 16) ----
  // One v_wmma_f32_16x16x32_f16 per 256-flag tile: A = [L|0] (16x32 lower-triangular
  // ones), B = [X;0] (32x16, sixteen contiguous 16-flag segments as columns).
  // ISA layouts: A lane m=ln&15 holds K = e(+8 if ln>=16) for e<8 (e>=8 -> K>=16 -> 0);
  // B lane n=ln&15 holds K = e(+8 if ln>=16) for e<8 (e>=8 -> 0); D lane n holds rows v(+8 if ln>=16).
  {
    const bool hi = (ln >= 16);
    const int  m  = ln & 15;
    v16h afrag, bfrag;
    for (int t = wv; t < NTILE_; t += 32) {
      for (int e = 0; e < 16; ++e) {
        const int ka = (e < 8) ? (e + (hi ? 8 : 0)) : 99;          // K index of A element
        afrag[e] = (ka <= m) ? (_Float16)1.0f : (_Float16)0.0f;    // L[m][ka]
        int v = 0;
        if (e < 8) {                                               // K = e(+8 if hi) < 16 rows hold X
          const int idx = 256 * t + m * 16 + e + (hi ? 8 : 0);     // column n = m here
          if (idx < K_TOT_) v = keep[idx];
        }
        bfrag[e] = (_Float16)(float)v;
      }
      v8f d = {};
#if defined(__HIP_DEVICE_COMPILE__)
      d = __builtin_amdgcn_wmma_f32_16x16x32_f16(false, afrag, false, bfrag,
                                                 (short)0, d, false, false);
#endif
      for (int v = 0; v < 8; ++v) {
        const int row = v + (hi ? 8 : 0);                          // M index
        rank[256 * t + m * 16 + row] = (int)d[v];                  // inclusive in-segment prefix
      }
    }
  }
  __syncthreads();

  if (tid == 0) {                      // cross-segment offsets (286 steps, was 4576)
    int off = 0;
    for (int s = 0; s < NSEG_; ++s) { segOff[s] = off; off += rank[s * 16 + 15]; }
  }
  __syncthreads();

  for (int q = tid; q < K_TOT_; q += nt) {
    if (keep[q]) {
      const int r = rank[q] - 1 + segOff[q >> 4];   // global exclusive rank among kept
      if (r < POST_) inv[r] = q;
    }
  }
  __syncthreads();

  for (int r = tid; r < POST_; r += nt) {
    float* o = out + ((size_t)b * POST_ + r) * 5;
    const int q = inv[r];
    if (q >= 0) {
      const uint32_t slot = order[(size_t)b * K_TOT_ + q];
      const float4 bx = candBox[(size_t)b * K_TOT_ + slot];  // clipped, un-offset box
      o[0] = bx.x; o[1] = bx.y; o[2] = bx.z; o[3] = bx.w;
      o[4] = candScore[(size_t)b * K_TOT_ + slot];
    } else {
      o[0] = 0.0f; o[1] = 0.0f; o[2] = 0.0f; o[3] = 0.0f; o[4] = 0.0f;
    }
  }
}

// ---------------- host launcher ----------------
extern "C" void kernel_launch(void* const* d_in, const int* in_sizes, int n_in,
                              void* d_out, int out_size, void* d_ws, size_t ws_size,
                              hipStream_t stream) {
  (void)in_sizes; (void)n_in; (void)out_size; (void)ws_size;
  const float* objectness = (const float*)d_in[0];   // [B, A]
  const float* deltas     = (const float*)d_in[1];   // [B, A, 4]
  const float* anchors    = (const float*)d_in[2];   // [A, 4]
  float* out = (float*)d_out;                        // [B, 1000, 5]

  char* ws = (char*)d_ws;
  float*    candScore = (float*)ws;     ws += (size_t)B_ * K_TOT_ * 4;   // 146,432
  uint32_t* candIdx   = (uint32_t*)ws;  ws += (size_t)B_ * K_TOT_ * 4;   // 146,432
  float4*   candBox   = (float4*)ws;    ws += (size_t)B_ * K_TOT_ * 16;  // 585,728
  uint32_t* order     = (uint32_t*)ws;  ws += (size_t)B_ * K_TOT_ * 4;   // 146,432
  float4*   sortedOff = (float4*)ws;    ws += (size_t)B_ * K_TOT_ * 16;  // 585,728
  uint32_t* maxBits   = (uint32_t*)ws;  ws += 64;                        // ~1.61 MB total

  k0_init<<<1, 32, 0, stream>>>(maxBits);
  k1_select_decode<<<40, 256, 0, stream>>>(objectness, deltas, anchors,
                                           candScore, candIdx, candBox, maxBits);
  const size_t sortShm = (size_t)SORT_N_ * 8 + (size_t)K_TOT_ * 8;       // 102,144 B
  k2_sort<<<B_, 1024, sortShm, stream>>>(candScore, candIdx, order);
  k3_materialize<<<dim3((K_TOT_ + 255) / 256, B_), 256, 0, stream>>>(order, candBox, maxBits, sortedOff);
  const size_t nmsShm = (size_t)K_TOT_ * 24 + (size_t)NTILE_ * 1024 + 288 * 4 + (size_t)POST_ * 4;
  k4_nms_out<<<B_, 1024, nmsShm, stream>>>(order, candScore, candBox, sortedOff, out);
}
